// MiniChessNNUE_42142219108846
// MI455X (gfx1250) — matrix operand order repaired and checked by reading, not compiled, not run on hardware
//
#include <hip/hip_runtime.h>

typedef __attribute__((ext_vector_type(16))) _Float16 v16h;
typedef __attribute__((ext_vector_type(8)))  float    v8f;

#define F_DIM   9000
#define ACC_DIM 128
#define KT      32
#define KSTEPS  ((F_DIM + KT - 1) / KT)   // 282 (last step zero-padded: 9000 mod 32 = 8)
#define BM      64                        // batch rows per workgroup

struct SmemMain {                         // double-buffered f16 staging tiles
  _Float16 wf[2][BM * KT];                // white features  64x32
  _Float16 bf[2][BM * KT];                // black features  64x32
  _Float16 w [2][ACC_DIM * KT];           // ft_w tile      128x32  ([n][k])
};                                        // 32 KB
struct SmemTail {
  _Float16 x  [BM * 256];                 // stm-ordered screlu accum  (32 KB)
  _Float16 l1h[32 * 256];                 // l1_w as f16 [n][k]        (16 KB)
  _Float16 l2h[32 * 32];                  // l2_w as f16 [n][k]        ( 2 KB)
  _Float16 x2 [BM * 32];                  // L1 activations            ( 4 KB)
  _Float16 x3 [BM * 32];                  // L2 activations            ( 4 KB)
  unsigned char smask[BM];
};
union Smem { SmemMain m; SmemTail t; };

__device__ __forceinline__ float screlu(float v) {
  float c = fminf(fmaxf(v, 0.0f), 1.0f);
  return c * c;
}

// A fragment (16xK=32 f16), per ISA layout: lanes 0-15 rows M=lane (k-base 0),
// lanes 16-31 same rows (k-base 8); VGPRs 0-3 = k..k+7, VGPRs 4-7 = k+16..k+23.
__device__ __forceinline__ v16h frag_a(const _Float16* p, int row, int stride, int kofs, int lane) {
  union { v16h v; uint4 q[2]; } u;
  const int kA = (lane & 16) ? 8 : 0;
  const _Float16* b = p + row * stride + kofs + kA;
  u.q[0] = *(const uint4*)(b);
  u.q[1] = *(const uint4*)(b + 16);
  return u.v;
}

// B fragment (K=32 x 16 f16): lanes 0-15 col N=lane, k 0..7,16..23; lanes 16-31 k 8..15,24..31.
// LDS stores B column-major-in-k, i.e. [n][k] — exactly ft_w's row-major layout.
__device__ __forceinline__ v16h frag_b(const _Float16* p, int col, int stride, int kofs, int lane) {
  union { v16h v; uint4 q[2]; } u;
  const int kB = (lane & 16) ? 8 : 0;
  const _Float16* b = p + col * stride + kofs + kB;
  u.q[0] = *(const uint4*)(b);
  u.q[1] = *(const uint4*)(b + 16);
  return u.v;
}

__global__ __launch_bounds__(256)
void nnue_fused(const float* __restrict__ wfeat, const float* __restrict__ bfeat,
                const unsigned char* __restrict__ stm,
                const float* __restrict__ ftw, const float* __restrict__ ftb,
                const float* __restrict__ l1w, const float* __restrict__ l1b,
                const float* __restrict__ l2w, const float* __restrict__ l2b,
                const float* __restrict__ outw, const float* __restrict__ outb,
                float* __restrict__ dout)
{
  __shared__ __align__(16) Smem s;
  const int tid  = threadIdx.x;
  const int lane = tid & 31;
  const int wid  = tid >> 5;            // 8 wave32s
  const int n16  = lane & 15;
  const int rowBase = blockIdx.x * BM;
  const int color = wid >> 2;           // 0 = white accum, 1 = black accum
  const int strip = wid & 3;            // 16-row M strip within the 64-row block

  float4 rg[8];                         // global->reg staging (8 float4 / thread)

  auto stage_load = [&](int ks) {
    const int k0 = ks * KT;
#pragma unroll
    for (int i = 0; i < 8; ++i) {
      const int c = tid + i * 256;      // 2048 float4 chunks per step
      const int col = (c & 7) << 2;     // k within tile (chunks of 4 floats)
      const float* src;
      if (c < 512)       src = wfeat + (size_t)(rowBase + (c >> 3)) * F_DIM;
      else if (c < 1024) src = bfeat + (size_t)(rowBase + ((c - 512) >> 3)) * F_DIM;
      else               src = ftw   + (size_t)((c - 1024) >> 3) * F_DIM;
      const int k = k0 + col;
      rg[i] = (k < F_DIM) ? *(const float4*)(src + k) : make_float4(0.f, 0.f, 0.f, 0.f);
    }
  };

  auto stage_store = [&](int buf) {
#pragma unroll
    for (int i = 0; i < 8; ++i) {
      const int c = tid + i * 256;
      const int col = (c & 7) << 2;
      _Float16* dst; int r;
      if (c < 512)       { r = c >> 3;          dst = s.m.wf[buf]; }
      else if (c < 1024) { r = (c - 512) >> 3;  dst = s.m.bf[buf]; }
      else               { r = (c - 1024) >> 3; dst = s.m.w[buf];  }
      union { _Float16 h[4]; uint2 u; } cv;
      cv.h[0] = (_Float16)rg[i].x; cv.h[1] = (_Float16)rg[i].y;
      cv.h[2] = (_Float16)rg[i].z; cv.h[3] = (_Float16)rg[i].w;
      *(uint2*)(dst + r * KT + col) = cv.u;
    }
  };

  v8f acc[8];
#pragma unroll
  for (int i = 0; i < 8; ++i) acc[i] = (v8f){0.f,0.f,0.f,0.f,0.f,0.f,0.f,0.f};

  stage_load(0);
  stage_store(0);
  __syncthreads();

  // ---- feature transform: (64 x 9000) @ (9000 x 128) for both colors ----
  for (int ks = 0; ks < KSTEPS; ++ks) {
    const int cur = ks & 1;
    const bool more = (ks + 1) < KSTEPS;
    if (more) stage_load(ks + 1);       // overlap next global loads with WMMA

    const _Float16* atile = color ? s.m.bf[cur] : s.m.wf[cur];
    const v16h af = frag_a(atile, strip * 16 + n16, KT, 0, lane);
#pragma unroll
    for (int nt = 0; nt < 8; ++nt) {
      const v16h bfr = frag_b(s.m.w[cur], nt * 16 + n16, KT, 0, lane);
      acc[nt] = __builtin_amdgcn_wmma_f32_16x16x32_f16(
          false, af, false, bfr, (short)0, acc[nt], false, false);
    }
    __syncthreads();
    if (more) { stage_store(cur ^ 1); __syncthreads(); }
  }

  // ---- tail constants into LDS (aliases main buffers; all reads done) ----
  if (tid < BM) s.t.smask[tid] = stm[rowBase + tid];
#pragma unroll
  for (int i = 0; i < 8; ++i) {         // l1_w: 8192 f32 -> f16 [32][256]
    const int c = tid + i * 256;
    const float4 f = *(const float4*)(l1w + c * 4);
    union { _Float16 h[4]; uint2 u; } cv;
    cv.h[0]=(_Float16)f.x; cv.h[1]=(_Float16)f.y; cv.h[2]=(_Float16)f.z; cv.h[3]=(_Float16)f.w;
    *(uint2*)(s.t.l1h + c * 4) = cv.u;
  }
  {                                     // l2_w: 1024 f32 -> f16 [32][32]
    const float4 f = *(const float4*)(l2w + tid * 4);
    union { _Float16 h[4]; uint2 u; } cv;
    cv.h[0]=(_Float16)f.x; cv.h[1]=(_Float16)f.y; cv.h[2]=(_Float16)f.z; cv.h[3]=(_Float16)f.w;
    *(uint2*)(s.t.l2h + tid * 4) = cv.u;
  }
  __syncthreads();

  // ---- bias + screlu + stm-select scatter into x[64][256] ----
  // C layout: VGPR r, lanes 0-15 -> M=r, lanes 16-31 -> M=r+8; N = lane%16.
#pragma unroll
  for (int nt = 0; nt < 8; ++nt) {
    const float bias = ftb[nt * 16 + n16];
#pragma unroll
    for (int r = 0; r < 8; ++r) {
      const int m  = strip * 16 + ((lane & 16) ? r + 8 : r);
      const int sm = s.t.smask[m] ? 1 : 0;
      const int cb = (sm == color) ? 0 : 128;   // cols 0-127 = stm accum
      s.t.x[m * 256 + cb + nt * 16 + n16] = (_Float16)screlu(acc[nt][r] + bias);
    }
  }
  __syncthreads();

  // ---- L1: (64x256)@(256x32), 8 waves = 4 strips x 2 N-tiles ----
  const int nt2 = wid >> 2;             // N tile 0..1
  v8f a1 = (v8f){0.f,0.f,0.f,0.f,0.f,0.f,0.f,0.f};
#pragma unroll
  for (int kc = 0; kc < 8; ++kc) {
    const v16h af  = frag_a(s.t.x,   strip * 16 + n16, 256, kc * 32, lane);
    const v16h bfr = frag_b(s.t.l1h, nt2   * 16 + n16, 256, kc * 32, lane);
    a1 = __builtin_amdgcn_wmma_f32_16x16x32_f16(false, af, false, bfr, (short)0, a1, false, false);
  }
  {
    const float bias = l1b[nt2 * 16 + n16];
#pragma unroll
    for (int r = 0; r < 8; ++r) {
      const int m = strip * 16 + ((lane & 16) ? r + 8 : r);
      s.t.x2[m * 32 + nt2 * 16 + n16] = (_Float16)screlu(a1[r] + bias);
    }
  }
  __syncthreads();

  // ---- L2: (64x32)@(32x32), one WMMA per wave ----
  v8f a2 = (v8f){0.f,0.f,0.f,0.f,0.f,0.f,0.f,0.f};
  {
    const v16h af  = frag_a(s.t.x2,  strip * 16 + n16, 32, 0, lane);
    const v16h bfr = frag_b(s.t.l2h, nt2   * 16 + n16, 32, 0, lane);
    a2 = __builtin_amdgcn_wmma_f32_16x16x32_f16(false, af, false, bfr, (short)0, a2, false, false);
    const float bias = l2b[nt2 * 16 + n16];
#pragma unroll
    for (int r = 0; r < 8; ++r) {
      const int m = strip * 16 + ((lane & 16) ? r + 8 : r);
      s.t.x3[m * 32 + nt2 * 16 + n16] = (_Float16)screlu(a2[r] + bias);
    }
  }
  __syncthreads();

  // ---- output head: 32-wide dot per row ----
  if (tid < BM) {
    float sum = outb[0];
#pragma unroll
    for (int j = 0; j < 32; ++j) sum += (float)s.t.x3[tid * 32 + j] * outw[j];
    dout[rowBase + tid] = sum;
  }
}

extern "C" void kernel_launch(void* const* d_in, const int* in_sizes, int n_in,
                              void* d_out, int out_size, void* d_ws, size_t ws_size,
                              hipStream_t stream) {
  const float* wf  = (const float*)d_in[0];
  const float* bf  = (const float*)d_in[1];
  const unsigned char* stm = (const unsigned char*)d_in[2];  // jax bool -> 1 byte
  const float* ftw = (const float*)d_in[3];
  const float* ftb = (const float*)d_in[4];
  const float* l1w = (const float*)d_in[5];
  const float* l1b = (const float*)d_in[6];
  const float* l2w = (const float*)d_in[7];
  const float* l2b = (const float*)d_in[8];
  const float* ow  = (const float*)d_in[9];
  const float* ob  = (const float*)d_in[10];
  float* out = (float*)d_out;

  const int rows = in_sizes[2];                   // B = 16384
  dim3 grid(rows / BM), block(256);
  hipLaunchKernelGGL(nnue_fused, grid, block, 0, stream,
                     wf, bf, stm, ftw, ftb, l1w, l1b, l2w, l2b, ow, ob, out);
}
